// My_emo_word_exam_81535659147880
// MI455X (gfx1250) — matrix operand order repaired and checked by reading, not compiled, not run on hardware
//
#include <hip/hip_runtime.h>
#include <math.h>

// ---------------------------------------------------------------------------
// MI455X (gfx1250) implementation.
// Heavy GEMMs: v_wmma_f32_16x16x32_bf16 (wave32 WMMA), fp32 accumulate,
// operands staged into LDS via the CDNA5 async copy engine
// (global_load_async_to_lds_b128, ASYNCcnt) with double buffering.
// B operands are kept N-major ("Bt") so every async chunk and every LDS
// fragment read is a contiguous 16-byte access.
// ---------------------------------------------------------------------------

typedef __bf16 bf16_t;
typedef __attribute__((ext_vector_type(16))) bf16_t v16bf;
typedef __attribute__((ext_vector_type(8)))  bf16_t v8bf;
typedef __attribute__((ext_vector_type(8)))  float  v8f;

#define B_   8
#define S_   1024
#define DIN_ 1024
#define DOUT_ 1024
#define P_   128
#define T_   9
#define M1_  (B_ * S_)   // 8192

#define KT_   32   // K elements per staged tile
#define APAD_ 40   // padded halfs per LDS row (32 data + 8 pad = 80B, debanked)

// ---------------------------------------------------------------------------
// fp32 -> bf16 convert (grid stride)
// ---------------------------------------------------------------------------
__global__ void cvt_f32_bf16(const float* __restrict__ in, bf16_t* __restrict__ out,
                             long long n) {
  long long i = (long long)blockIdx.x * blockDim.x + threadIdx.x;
  long long stride = (long long)gridDim.x * blockDim.x;
  for (; i < n; i += stride) out[i] = (bf16_t)in[i];
}

// ---------------------------------------------------------------------------
// Tiled transpose to bf16: out[col][row] = (bf16)in[row][col]
// blockDim = (32, 8); grid = (cols/32, rows/32, batch)
// ---------------------------------------------------------------------------
template <typename T>
__global__ void transpose_to_bf16(const T* __restrict__ in, bf16_t* __restrict__ out,
                                  int rows, int cols, long long sIn, long long sOut) {
  __shared__ float tile[32][33];
  in  += (long long)blockIdx.z * sIn;
  out += (long long)blockIdx.z * sOut;
  int c0 = blockIdx.x * 32, r0 = blockIdx.y * 32;
  int tx = threadIdx.x, ty = threadIdx.y;
#pragma unroll
  for (int dy = 0; dy < 32; dy += 8) {
    int r = r0 + ty + dy, c = c0 + tx;
    tile[ty + dy][tx] = (float)in[(long long)r * cols + c];
  }
  __syncthreads();
#pragma unroll
  for (int dy = 0; dy < 32; dy += 8) {
    int col = c0 + ty + dy, row = r0 + tx;
    out[(long long)col * rows + row] = (bf16_t)tile[tx][ty + dy];
  }
}

// ---------------------------------------------------------------------------
// WMMA GEMM with async-LDS double buffering.
//   C[m,n] = act( sum_k A[m,k] * Bt[n,k] + bias[n] )
//   A  : [M x K] bf16 row-major,  Bt : [N x K] bf16 (B column-major)
// Block = 128 threads (4 waves). Block tile 128M x 64N; each wave owns a
// 32M x 64N slice (2x4 WMMA tiles), all waves share the LDS B panel.
// Per K-tile (32) each wave issues exactly 6 global_load_async_to_lds_b128:
// 4 for the 128x32 A panel, 2 for the 64x32 B panel. Double buffered with
// s_wait_asynccnt 6 (async loads complete in order).
// grid = (N/64, M/128, batch)
// ---------------------------------------------------------------------------
template <bool RELU, bool BIAS, bool OUT_BF16>
__global__ void gemm_wmma_bf16(const bf16_t* __restrict__ A,
                               const bf16_t* __restrict__ Bt,
                               const float* __restrict__ bias,
                               float* __restrict__ Cf, bf16_t* __restrict__ Cb,
                               int M, int N, int K,
                               long long sA, long long sB, long long sC) {
  __shared__ __align__(16) bf16_t shA[2][128][APAD_];
  __shared__ __align__(16) bf16_t shB[2][64][APAD_];

  const int tid  = threadIdx.x;
  const int lane = tid & 31;
  const int wave = tid >> 5;
  const int bz   = blockIdx.z;
  A  += (long long)bz * sA;
  Bt += (long long)bz * sB;
  const long long cbase = (long long)bz * sC;

  const int n0 = blockIdx.x * 64;    // block N tile
  const int mB = blockIdx.y * 128;   // block M tile
  const int m0 = mB + wave * 32;     // this wave's M slice

  const int half = lane >> 4;        // 0: lanes 0-15, 1: lanes 16-31
  const int lr   = lane & 15;

  // Issue one K-tile's async copies (6 wave-instructions per wave).
  auto issue_tile = [&](int buf, int k0) {
    // A panel: 128 rows x 32 halfs; 16B chunks -> 512 items, 4 per thread.
#pragma unroll
    for (int c = 0; c < 4; ++c) {
      int item = c * 128 + tid;
      int r = item >> 2, ch = item & 3;
      const bf16_t* g = A + (long long)(mB + r) * K + k0 + ch * 8;
      unsigned l = (unsigned)(size_t)(&shA[buf][r][ch * 8]);
      asm volatile("global_load_async_to_lds_b128 %0, %1, off"
                   :: "v"(l), "v"(g) : "memory");
    }
    // B panel: 64 rows x 32 halfs -> 256 items, 2 per thread.
#pragma unroll
    for (int c = 0; c < 2; ++c) {
      int item = c * 128 + tid;
      int r = item >> 2, ch = item & 3;
      const bf16_t* g = Bt + (long long)(n0 + r) * K + k0 + ch * 8;
      unsigned l = (unsigned)(size_t)(&shB[buf][r][ch * 8]);
      asm volatile("global_load_async_to_lds_b128 %0, %1, off"
                   :: "v"(l), "v"(g) : "memory");
    }
  };

  v8f acc[2][4] = {};
  const int NT = K / KT_;

  issue_tile(0, 0);
  for (int it = 0; it < NT; ++it) {
    if (it + 1 < NT) {
      issue_tile((it + 1) & 1, (it + 1) * KT_);
      asm volatile("s_wait_asynccnt 0x6" ::: "memory");  // tile `it` landed
    } else {
      asm volatile("s_wait_asynccnt 0x0" ::: "memory");
    }
    __syncthreads();  // all waves' portions of tile `it` visible

    const int buf = it & 1;
    // A fragments: lane holds row m0+i*16+lr, K halves {half*8..+7},{16+half*8..+7}
    v16bf afrag[2];
#pragma unroll
    for (int i = 0; i < 2; ++i) {
      const bf16_t* ap = &shA[buf][wave * 32 + i * 16 + lr][half * 8];
      v8bf lo = *(const v8bf*)(ap);
      v8bf hi = *(const v8bf*)(ap + 16);
      afrag[i] = __builtin_shufflevector(lo, hi, 0, 1, 2, 3, 4, 5, 6, 7,
                                         8, 9, 10, 11, 12, 13, 14, 15);
    }
#pragma unroll
    for (int j = 0; j < 4; ++j) {
      // B fragment: lane holds column n0+j*16+lr, K halves {half*8..+7},{16+half*8..+7}
      const bf16_t* bp = &shB[buf][j * 16 + lr][half * 8];
      v8bf b0 = *(const v8bf*)(bp);
      v8bf b1 = *(const v8bf*)(bp + 16);
      v16bf bfrag = __builtin_shufflevector(b0, b1, 0, 1, 2, 3, 4, 5, 6, 7,
                                            8, 9, 10, 11, 12, 13, 14, 15);
#pragma unroll
      for (int i = 0; i < 2; ++i) {
        acc[i][j] = __builtin_amdgcn_wmma_f32_16x16x32_bf16(
            false, afrag[i], false, bfrag, (short)0, acc[i][j], false, false);
      }
    }
    __syncthreads();  // tile consumed; its buffer may be overwritten next issue
  }

  // Store: per 16x16 tile, lane holds D rows M = v + 8*half, column N = lr
#pragma unroll
  for (int j = 0; j < 4; ++j) {
    int n = n0 + j * 16 + lr;
    float bv = BIAS ? bias[n] : 0.0f;
#pragma unroll
    for (int i = 0; i < 2; ++i) {
#pragma unroll
      for (int v = 0; v < 8; ++v) {
        int m = m0 + i * 16 + v + half * 8;
        float val = acc[i][j][v] + bv;
        if (RELU) val = fmaxf(val, 0.0f);
        long long idx = cbase + (long long)m * N + n;
        if (OUT_BF16) Cb[idx] = (bf16_t)val;
        else          Cf[idx] = val;
      }
    }
  }
}

// ---------------------------------------------------------------------------
// Row softmax: one 256-thread block per row of length n; f32 in, bf16 out.
// ---------------------------------------------------------------------------
__global__ void softmax_row(const float* __restrict__ L, bf16_t* __restrict__ att, int n) {
  long long row = blockIdx.x;
  const float* x = L + row * n;
  bf16_t* y = att + row * n;
  __shared__ float red[256];
  int t = threadIdx.x;

  float m = -3.4e38f;
  for (int i = t; i < n; i += 256) m = fmaxf(m, x[i]);
  red[t] = m; __syncthreads();
  for (int s = 128; s > 0; s >>= 1) {
    if (t < s) red[t] = fmaxf(red[t], red[t + s]);
    __syncthreads();
  }
  m = red[0]; __syncthreads();

  float sum = 0.0f;
  for (int i = t; i < n; i += 256) sum += expf(x[i] - m);
  red[t] = sum; __syncthreads();
  for (int s = 128; s > 0; s >>= 1) {
    if (t < s) red[t] += red[t + s];
    __syncthreads();
  }
  float inv = 1.0f / red[0];

  for (int i = t; i < n; i += 256) y[i] = (bf16_t)(expf(x[i] - m) * inv);
}

// ---------------------------------------------------------------------------
// Emissions: em[b,s,t] = b_hid[t] + p_cla[b,s,:]@W_hid[:128,t] + out2[b,s,:]@W_hid[128:,t]
// ---------------------------------------------------------------------------
__global__ void em_kernel(const float* __restrict__ p_cla, const bf16_t* __restrict__ out2,
                          const float* __restrict__ W_hid, const float* __restrict__ b_hid,
                          float* __restrict__ em) {
  int idx = blockIdx.x * blockDim.x + threadIdx.x;
  if (idx >= B_ * S_ * T_) return;
  int t = idx % T_;
  int bs = idx / T_;
  const float*  pc = p_cla + (long long)bs * P_;
  const bf16_t* o  = out2 + (long long)bs * DOUT_;
  float acc = b_hid[t];
  for (int p = 0; p < P_; ++p) acc += pc[p] * W_hid[p * T_ + t];
  for (int d = 0; d < DOUT_; ++d) acc += (float)o[d] * W_hid[(P_ + d) * T_ + t];
  em[idx] = acc;
}

// ---------------------------------------------------------------------------
// CRF NLL: single 128-thread block. 72 live lanes run the forward recursion;
// 8 lanes accumulate the numerator. out[0] = sum_b(logZ_b - num_b).
// ---------------------------------------------------------------------------
__global__ void crf_kernel(const float* __restrict__ em, const int* __restrict__ tags,
                           const float* __restrict__ start, const float* __restrict__ endv,
                           const float* __restrict__ trans, float* __restrict__ out) {
  __shared__ float s_tr[T_][T_];
  __shared__ float alpha[2][B_][T_];
  __shared__ float s_logZ[B_], s_num[B_];
  int t = threadIdx.x;
  if (t < T_ * T_) s_tr[t / T_][t % T_] = trans[t];
  if (t < B_ * T_) {
    int b = t / T_, j = t % T_;
    alpha[0][b][j] = start[j] + em[(long long)(b * S_) * T_ + j];
  }
  __syncthreads();

  int cur = 0;
  for (int s = 1; s < S_; ++s) {
    if (t < B_ * T_) {
      int b = t / T_, j = t % T_;
      float mx = -3.4e38f;
#pragma unroll
      for (int i = 0; i < T_; ++i) mx = fmaxf(mx, alpha[cur][b][i] + s_tr[i][j]);
      float sum = 0.0f;
#pragma unroll
      for (int i = 0; i < T_; ++i) sum += expf(alpha[cur][b][i] + s_tr[i][j] - mx);
      alpha[1 - cur][b][j] = em[((long long)b * S_ + s) * T_ + j] + mx + logf(sum);
    }
    cur = 1 - cur;
    __syncthreads();
  }

  if (t < B_) {
    int b = t;
    float mx = -3.4e38f;
    for (int j = 0; j < T_; ++j) mx = fmaxf(mx, alpha[cur][b][j] + endv[j]);
    float sum = 0.0f;
    for (int j = 0; j < T_; ++j) sum += expf(alpha[cur][b][j] + endv[j] - mx);
    s_logZ[b] = mx + logf(sum);

    int prev = tags[b * S_];
    float num = start[prev] + em[((long long)b * S_) * T_ + prev];
    for (int s = 1; s < S_; ++s) {
      int tg = tags[b * S_ + s];
      num += em[((long long)b * S_ + s) * T_ + tg] + s_tr[prev][tg];
      prev = tg;
    }
    num += endv[prev];
    s_num[b] = num;
  }
  __syncthreads();
  if (t == 0) {
    float r = 0.0f;
    for (int b = 0; b < B_; ++b) r += s_logZ[b] - s_num[b];
    out[0] = r;
  }
}

// ---------------------------------------------------------------------------
// Host launcher
// ---------------------------------------------------------------------------
extern "C" void kernel_launch(void* const* d_in, const int* in_sizes, int n_in,
                              void* d_out, int out_size, void* d_ws, size_t ws_size,
                              hipStream_t stream) {
  const float* x        = (const float*)d_in[0];
  const float* p_cla    = (const float*)d_in[1];
  const float* W_emo    = (const float*)d_in[2];
  const float* b_emo    = (const float*)d_in[3];
  const float* W_att    = (const float*)d_in[4];
  const float* b_att    = (const float*)d_in[5];
  const float* W_hid    = (const float*)d_in[6];
  const float* b_hid    = (const float*)d_in[7];
  const float* crf_start= (const float*)d_in[8];
  const float* crf_end  = (const float*)d_in[9];
  const float* crf_trans= (const float*)d_in[10];
  const int*   emo_list = (const int*)d_in[11];
  float* outp = (float*)d_out;

  // Workspace carve-out (256B aligned)
  char* ws = (char*)d_ws;
  size_t off = 0;
  auto carve = [&](size_t bytes) -> char* {
    char* p = ws + off;
    off = (off + bytes + 255) & ~(size_t)255;
    return p;
  };
  bf16_t* Xbf   = (bf16_t*)carve((size_t)M1_ * DIN_ * 2);    // 16 MB (reused as Ht later)
  bf16_t* WemoT = (bf16_t*)carve((size_t)DIN_ * DOUT_ * 2);  // 2 MB
  bf16_t* WattT = (bf16_t*)carve((size_t)DOUT_ * DOUT_ * 2); // 2 MB
  bf16_t* Hbf   = (bf16_t*)carve((size_t)M1_ * DOUT_ * 2);   // 16 MB
  bf16_t* HAbf  = (bf16_t*)carve((size_t)M1_ * DOUT_ * 2);   // 16 MB
  float*  Lf    = (float*) carve((size_t)B_ * S_ * S_ * 4);  // 33.5 MB (reused for out2)
  bf16_t* attbf = (bf16_t*)carve((size_t)B_ * S_ * S_ * 2);  // 16 MB
  float*  em    = (float*) carve((size_t)B_ * S_ * T_ * 4);  // 0.3 MB
  bf16_t* Htbf  = Xbf;            // X dead after GEMM1
  bf16_t* out2  = (bf16_t*)Lf;    // logits dead after softmax
  (void)in_sizes; (void)n_in; (void)out_size; (void)ws_size;

  // 1) fp32 -> bf16 for X
  cvt_f32_bf16<<<4096, 256, 0, stream>>>(x, Xbf, (long long)M1_ * DIN_);

  // 2) Transposing converts for weights: Wt[n][k] = W[k][n]
  {
    dim3 blk(32, 8);
    dim3 grd(DOUT_ / 32, DIN_ / 32, 1);
    transpose_to_bf16<float><<<grd, blk, 0, stream>>>(W_emo, WemoT, DIN_, DOUT_, 0, 0);
    transpose_to_bf16<float><<<grd, blk, 0, stream>>>(W_att, WattT, DOUT_, DOUT_, 0, 0);
  }

  // 3) H = relu(X @ W_emo + b_emo)   [8192 x 1024]
  gemm_wmma_bf16<true, true, true><<<dim3(DOUT_ / 64, M1_ / 128, 1), 128, 0, stream>>>(
      Xbf, WemoT, b_emo, nullptr, Hbf, M1_, DOUT_, DIN_, 0, 0, 0);

  // 4) HA = H @ W_att + b_att        [8192 x 1024]
  gemm_wmma_bf16<false, true, true><<<dim3(DOUT_ / 64, M1_ / 128, 1), 128, 0, stream>>>(
      Hbf, WattT, b_att, nullptr, HAbf, M1_, DOUT_, DOUT_, 0, 0, 0);

  // 5) Per batch: L = HA_b @ H_b^T  (H row-major already serves as Bt)
  gemm_wmma_bf16<false, false, false><<<dim3(S_ / 64, S_ / 128, B_), 128, 0, stream>>>(
      HAbf, Hbf, nullptr, Lf, nullptr, S_, S_, DOUT_,
      (long long)S_ * DOUT_, (long long)S_ * DOUT_, (long long)S_ * S_);

  // 6) att = softmax_rows(L) -> bf16
  softmax_row<<<B_ * S_, 256, 0, stream>>>(Lf, attbf, S_);

  // 7) Ht_b[d][t] = H_b[t][d] (bf16 tile transpose, per batch)
  {
    dim3 blk(32, 8);
    dim3 grd(DOUT_ / 32, S_ / 32, B_);
    transpose_to_bf16<bf16_t><<<grd, blk, 0, stream>>>(
        Hbf, Htbf, S_, DOUT_, (long long)S_ * DOUT_, (long long)S_ * DOUT_);
  }

  // 8) out2 = att_b @ H_b  (Bt = Ht_b)
  gemm_wmma_bf16<false, false, true><<<dim3(DOUT_ / 64, S_ / 128, B_), 128, 0, stream>>>(
      attbf, Htbf, nullptr, nullptr, out2, S_, DOUT_, S_,
      (long long)S_ * S_, (long long)S_ * DOUT_, (long long)S_ * DOUT_);

  // 9) em = [p_cla | out2] @ W_hid + b_hid   [B,S,9]
  em_kernel<<<(B_ * S_ * T_ + 255) / 256, 256, 0, stream>>>(p_cla, out2, W_hid, b_hid, em);

  // 10) CRF negative log-likelihood -> scalar
  crf_kernel<<<1, 128, 0, stream>>>(em, emo_list, crf_start, crf_end, crf_trans, outp);
}
